// PositLayerNet_56109452755006
// MI455X (gfx1250) — compile-verified, matrix operand order checked by reading, not run on hardware
//
#include <hip/hip_runtime.h>
#include <hip/hip_bf16.h>

// ---------------------------------------------------------------------------
// Types for WMMA fragments (wave32, gfx1250)
// ---------------------------------------------------------------------------
typedef __attribute__((ext_vector_type(16))) _Float16 v16h;
typedef __attribute__((ext_vector_type(8)))  _Float16 v8h;
typedef __attribute__((ext_vector_type(4)))  _Float16 v4h;
typedef __attribute__((ext_vector_type(8)))  float    v8f;

union Frag16 {
    v16h v;
    v8h  h[2];
};

static constexpr int LDSK    = 40;           // padded K-pitch in halves (80 B rows)
static constexpr int TILE_HB = 128 * LDSK;   // halves per (128 x 32) tile buffer

// Convert a generic pointer to a shared-memory object into the 32-bit
// wave-relative LDS byte address that async-to-LDS instructions consume.
__device__ __forceinline__ unsigned lds_addr_of(const void* p)
{
    return (unsigned)(unsigned long long)
        (const __attribute__((address_space(3))) void*)p;
}

// CDNA5 async DMA: global -> LDS, 16 bytes per lane, tracked by ASYNCcnt.
__device__ __forceinline__ void async_load_b128(unsigned lds_byte_addr, const void* gptr)
{
    asm volatile("global_load_async_to_lds_b128 %0, %1, off"
                 :: "v"(lds_byte_addr), "v"(gptr)
                 : "memory");
}

// ---------------------------------------------------------------------------
// posit(16, es=1) round-trip quantizer, matching the JAX reference:
//   scale  = floor(log2(|x|)), regime = floor(scale/2)   (useed = 2^2)
//   rlen   = regime>=0 ? regime+2 : 1-regime
//   fbits  = clip(14 - rlen, 0, 28); RNE-round mantissa to fbits
//   clamp magnitude to [2^-28, 2^28]; sign(0) = 0
// ---------------------------------------------------------------------------
__device__ __forceinline__ float posit_quant(float x)
{
    float ax   = __builtin_fabsf(x);
    float safe = fmaxf(ax, 1e-30f);
    float scl  = floorf(log2f(safe));
    int   e    = (int)scl;
    float reg  = floorf(scl * 0.5f);
    float rlen = (reg >= 0.0f) ? (reg + 2.0f) : (1.0f - reg);
    float fb   = fminf(fmaxf(14.0f - rlen, 0.0f), 28.0f);
    float pw   = ldexpf(1.0f, (int)fb);
    float m    = ldexpf(safe, -e);                    // mantissa in [1,2), exact
    float mq   = 1.0f + rintf((m - 1.0f) * pw) / pw;  // RNE like jnp.round
    float mag  = fminf(fmaxf(ldexpf(mq, e), 0x1p-28f), 0x1p28f);
    float q    = (ax == 0.0f) ? 0.0f : copysignf(mag, x);
    return q;
}

// ---------------------------------------------------------------------------
// Elementwise pre-pass: f32 -> posit-quantize -> f16 (feeds the WMMA GEMMs)
// ---------------------------------------------------------------------------
__global__ __launch_bounds__(256) void posit_quant_to_f16(
    const float* __restrict__ in, _Float16* __restrict__ out, int n)
{
    int i = (blockIdx.x * 256 + threadIdx.x) * 4;
    if (i < n) {
        float4 v = *(const float4*)(in + i);
        v4h o;
        o[0] = (_Float16)posit_quant(v.x);
        o[1] = (_Float16)posit_quant(v.y);
        o[2] = (_Float16)posit_quant(v.z);
        o[3] = (_Float16)posit_quant(v.w);
        *(v4h*)(out + i) = o;
    }
}

// ---------------------------------------------------------------------------
// WMMA GEMM:  out[M,N] = epilogue( A[M,K] @ Bw[N,K]^T + bias[N] )
//   A, Bw are posit-quantized f16. f32 accumulate via v_wmma_f32_16x16x32_f16.
//   Epilogue: acc + posit_quant(bias) -> posit_quant -> (optional ReLU).
//
// Block: 256 threads = 8 waves; tile 128(M) x 128(N); K-step 32.
// Waves 4(M) x 2(N); each wave owns 2x4 WMMA 16x16 tiles (8 wmma / K-step).
// Double-buffered LDS (40 KB) filled by global_load_async_to_lds_b128.
// Steady-state loop is branch-free (last K-step peeled): stage tile kt+1,
// s_wait_asynccnt 4 (in-order ASYNCcnt retires exactly the current tile),
// barrier, fragment ds_loads + 8 WMMA, s_wait_dscnt 0, barrier.
// Per-thread global chunk pointers are hoisted and bumped 64 B per step.
// All dims divide the tile sizes -> no bounds checks; EXEC is all-ones
// wherever WMMA executes (ISA requirement).
// ---------------------------------------------------------------------------
template<bool RELU, bool OUTF16>
__global__ __launch_bounds__(256) void posit_gemm_wmma(
    const _Float16* __restrict__ A,
    const _Float16* __restrict__ Bw,
    const float*    __restrict__ bias,
    void*           __restrict__ outp,
    int M, int N, int K)
{
    __shared__ __attribute__((aligned(16))) _Float16 sA[2][TILE_HB];
    __shared__ __attribute__((aligned(16))) _Float16 sB[2][TILE_HB];

    const int tid   = threadIdx.x;
    const int lane  = tid & 31;
    const int wave  = tid >> 5;
    const int waveM = wave & 3;          // 0..3  -> 4*32 = 128 rows
    const int waveN = wave >> 2;         // 0..1  -> 2*64 = 128 cols
    const int l16   = lane & 15;
    const int hi    = (lane >> 4) & 1;
    const int kb    = hi ? 8 : 0;        // fragment K-chunk base per ISA layout

    const int mBase = blockIdx.y * 128;
    const int nBase = blockIdx.x * 128;

    // ---- hoisted per-thread staging addresses (2 chunks per operand) ----
    const int r0 = tid >> 2,          c0 = (tid & 3) << 3;          // chunk tid
    const int r1 = (tid + 256) >> 2,  c1 = (tid & 3) << 3;          // chunk tid+256
    const _Float16* gA0 = A  + (size_t)(mBase + r0) * K + c0;
    const _Float16* gA1 = A  + (size_t)(mBase + r1) * K + c1;
    const _Float16* gB0 = Bw + (size_t)(nBase + r0) * K + c0;
    const _Float16* gB1 = Bw + (size_t)(nBase + r1) * K + c1;
    unsigned lA0[2], lA1[2], lB0[2], lB1[2];
#pragma unroll
    for (int bbuf = 0; bbuf < 2; ++bbuf) {
        lA0[bbuf] = lds_addr_of(&sA[bbuf][r0 * LDSK + c0]);
        lA1[bbuf] = lds_addr_of(&sA[bbuf][r1 * LDSK + c1]);
        lB0[bbuf] = lds_addr_of(&sB[bbuf][r0 * LDSK + c0]);
        lB1[bbuf] = lds_addr_of(&sB[bbuf][r1 * LDSK + c1]);
    }

    // Issue one 128x32 A tile + 128x32 B tile (4 async ops / lane / stage),
    // then advance the global pointers to the next K-slab.
    auto stage = [&](int bbuf) {
        async_load_b128(lA0[bbuf], gA0);
        async_load_b128(lB0[bbuf], gB0);
        async_load_b128(lA1[bbuf], gA1);
        async_load_b128(lB1[bbuf], gB1);
        gA0 += 32; gA1 += 32; gB0 += 32; gB1 += 32;
    };

    v8f acc[2][4] = {};

    // One K-step of compute from LDS buffer `bbuf`.
    auto compute = [&](int bbuf) {
        const _Float16* cA = sA[bbuf];
        const _Float16* cB = sB[bbuf];
        Frag16 fa[2], fb[4];
#pragma unroll
        for (int mi = 0; mi < 2; ++mi) {
            int r = waveM * 32 + mi * 16 + l16;
            fa[mi].h[0] = *(const v8h*)&cA[r * LDSK + kb];
            fa[mi].h[1] = *(const v8h*)&cA[r * LDSK + kb + 16];
        }
#pragma unroll
        for (int ni = 0; ni < 4; ++ni) {
            int r = waveN * 64 + ni * 16 + l16;
            fb[ni].h[0] = *(const v8h*)&cB[r * LDSK + kb];
            fb[ni].h[1] = *(const v8h*)&cB[r * LDSK + kb + 16];
        }
#pragma unroll
        for (int mi = 0; mi < 2; ++mi)
#pragma unroll
            for (int ni = 0; ni < 4; ++ni)
                acc[mi][ni] = __builtin_amdgcn_wmma_f32_16x16x32_f16(
                    /*neg_a=*/false, fa[mi].v,
                    /*neg_b=*/false, fb[ni].v,
                    /*c_mod=*/(short)0, acc[mi][ni],
                    /*reuse_a=*/false, /*reuse_b=*/false);
    };

    const int nk = K >> 5;               // K-steps of 32 (>= 64 for this net)

    // Prologue: DMA tile 0 into buffer 0.
    stage(0);

    // Steady state (branch-free body): tiles 0 .. nk-2.
    for (int kt = 0; kt < nk - 1; ++kt) {
        const int b = kt & 1;
        stage(b ^ 1);                                   // DMA tile kt+1
        asm volatile("s_wait_asynccnt 0x4" ::: "memory"); // tile kt retired
        __syncthreads();                                // visible to all waves
        compute(b);
        asm volatile("s_wait_dscnt 0x0" ::: "memory");  // my LDS reads done
        __syncthreads();                                // safe to overwrite buf b
    }

    // Peeled last K-step: nothing left to stage.
    asm volatile("s_wait_asynccnt 0x0" ::: "memory");
    __syncthreads();
    compute((nk - 1) & 1);

    // ---- epilogue: +posit(bias), posit-quantize output, optional ReLU ----
    // C layout (32-bit 16x16): VGPR j holds M = j + 8*hi, N = lane%16.
#pragma unroll
    for (int ni = 0; ni < 4; ++ni) {
        const int col = nBase + waveN * 64 + ni * 16 + l16;
        const float bq = posit_quant(bias[col]);
#pragma unroll
        for (int mi = 0; mi < 2; ++mi) {
#pragma unroll
            for (int j = 0; j < 8; ++j) {
                int row = mBase + waveM * 32 + mi * 16 + hi * 8 + j;
                float v = posit_quant(acc[mi][ni][j] + bq);
                if (RELU) v = fmaxf(v, 0.0f);
                if (OUTF16)
                    ((_Float16*)outp)[(size_t)row * N + col] = (_Float16)v;
                else
                    ((float*)outp)[(size_t)row * N + col] = v;
            }
        }
    }
}

// ---------------------------------------------------------------------------
// Host launch
//   inputs: x[4096,2048], W0[8192,2048], b0[8192], W1[8192,8192], b1[8192],
//           W2[2048,8192], b2[2048]   (all f32)
//   output: [4096,2048] f32
// Workspace layout (f16 buffers), ~336 MiB total.
// ---------------------------------------------------------------------------
static constexpr int    Mdim = 4096;
static constexpr int    K0   = 2048, N0 = 8192, N1 = 8192, N2 = 2048;
static constexpr size_t OFF_XQ  = 0;
static constexpr size_t OFF_W0Q = OFF_XQ  + (size_t)Mdim * K0 * 2;   //  16 MiB
static constexpr size_t OFF_W1Q = OFF_W0Q + (size_t)N0 * K0 * 2;     // +32 MiB
static constexpr size_t OFF_W2Q = OFF_W1Q + (size_t)N1 * N0 * 2;     // +128 MiB
static constexpr size_t OFF_H0  = OFF_W2Q + (size_t)N2 * N1 * 2;     // +32 MiB
static constexpr size_t OFF_H1  = OFF_H0  + (size_t)Mdim * N0 * 2;   // +64 MiB

extern "C" void kernel_launch(void* const* d_in, const int* in_sizes, int n_in,
                              void* d_out, int out_size, void* d_ws, size_t ws_size,
                              hipStream_t stream)
{
    (void)in_sizes; (void)n_in; (void)out_size; (void)ws_size;

    const float* x  = (const float*)d_in[0];
    const float* W0 = (const float*)d_in[1];
    const float* b0 = (const float*)d_in[2];
    const float* W1 = (const float*)d_in[3];
    const float* b1 = (const float*)d_in[4];
    const float* W2 = (const float*)d_in[5];
    const float* b2 = (const float*)d_in[6];
    float* out = (float*)d_out;

    char* ws = (char*)d_ws;
    _Float16* xq  = (_Float16*)(ws + OFF_XQ);
    _Float16* W0q = (_Float16*)(ws + OFF_W0Q);
    _Float16* W1q = (_Float16*)(ws + OFF_W1Q);
    _Float16* W2q = (_Float16*)(ws + OFF_W2Q);
    _Float16* h0  = (_Float16*)(ws + OFF_H0);
    _Float16* h1  = (_Float16*)(ws + OFF_H1);

    auto quant = [&](const float* src, _Float16* dst, size_t n) {
        int blocks = (int)((n / 4 + 255) / 256);
        posit_quant_to_f16<<<blocks, 256, 0, stream>>>(src, dst, (int)n);
    };

    // Posit-quantize operands once into f16 (W1q = 128 MiB -> lives in 192 MB L2)
    quant(x,  xq,  (size_t)Mdim * K0);
    quant(W0, W0q, (size_t)N0 * K0);
    quant(W1, W1q, (size_t)N1 * N0);
    quant(W2, W2q, (size_t)N2 * N1);

    // Layer 0: h0 = relu(q(xq @ W0q^T + q(b0)))      [4096 x 8192] f16
    posit_gemm_wmma<true,  true ><<<dim3(N0 / 128, Mdim / 128), 256, 0, stream>>>(
        xq, W0q, b0, h0, Mdim, N0, K0);

    // Layer 1: h1 = relu(q(h0 @ W1q^T + q(b1)))      [4096 x 8192] f16
    posit_gemm_wmma<true,  true ><<<dim3(N1 / 128, Mdim / 128), 256, 0, stream>>>(
        h0, W1q, b1, h1, Mdim, N1, N0);

    // Layer 2: out = q(h1 @ W2q^T + q(b2))           [4096 x 2048] f32
    posit_gemm_wmma<false, false><<<dim3(N2 / 128, Mdim / 128), 256, 0, stream>>>(
        h1, W2q, b2, out, Mdim, N2, N1);
}
